// ET_57123065036972
// MI455X (gfx1250) — matrix operand order repaired and checked by reading, not compiled, not run on hardware
//
#include <hip/hip_runtime.h>

// ---------------------------------------------------------------------------
// Energy Transformer forward on MI455X (gfx1250, wave32, WMMA bf16).
// All heavy math runs through v_wmma_f32_16x16x32_bf16.
// Token axis padded 197 -> 224; every GEMM N is a multiple of 64 so the
// inner loop is branch-free. Each wave computes a 32(M) x 64(N) tile.
// Fusions: [Q|K] projection in one GEMM (N=1536); the three energy-gradient
// terms are written into one ACT[M,4608] buffer and applied to t with a
// single fused accumulate GEMM against CW = [wq^T | wk^T | w_hop^T].
// ---------------------------------------------------------------------------

typedef __attribute__((ext_vector_type(16))) __bf16 v16bf;
typedef __attribute__((ext_vector_type(8)))  float  v8f;

#define DEVI __device__ __forceinline__

constexpr int  B_   = 32;
constexpr int  N_   = 196;
constexpr int  NB   = 197;   // valid tokens (cls + 196)
constexpr int  NP   = 224;   // padded tokens (multiple of 32)
constexpr int  SLD  = 256;   // score-matrix leading dim (multiple of 64)
constexpr int  D    = 768;   // patch dim
constexpr int  TKN  = 768;
constexpr int  H    = 12;
constexpr int  QK   = 64;
constexpr int  HID  = 3072;
constexpr int  KST  = 12;
constexpr int  QKLD = 2 * TKN;          // 1536: combined [Q|K] leading dim
constexpr int  AC   = 2 * TKN + HID;    // 4608: combined activation width
constexpr float BETA = 0.125f;   // 1/sqrt(64)
constexpr float EPS  = 1e-5f;

DEVI unsigned short f2bf(float f) {
  unsigned int u = __float_as_uint(f);
  u += 0x7fffu + ((u >> 16) & 1u);          // round-to-nearest-even
  return (unsigned short)(u >> 16);
}

union FragB16 {
  uint4          q[2];
  unsigned short s[16];
  v16bf          v;
};

// ---------------------------------------------------------------------------
// fp32 -> bf16 conversion (plain / transposing-into-strided-destination)
// ---------------------------------------------------------------------------
__global__ void cvt_bf16(const float* __restrict__ src,
                         unsigned short* __restrict__ dst, long n) {
  long i = (long)blockIdx.x * blockDim.x + threadIdx.x;
  long st = (long)gridDim.x * blockDim.x;
  for (; i < n; i += st) dst[i] = f2bf(src[i]);
}

// dst[c*dld + coff + r] = src[r*C + c]  (src is R x C row-major)
__global__ void cvt_bf16_t_ld(const float* __restrict__ src,
                              unsigned short* __restrict__ dst,
                              int R, int C, int dld, int coff) {
  long n = (long)R * C;
  long i = (long)blockIdx.x * blockDim.x + threadIdx.x;
  long st = (long)gridDim.x * blockDim.x;
  for (; i < n; i += st) {
    long r = i / C, c = i - r * C;
    dst[c * (long)dld + coff + r] = f2bf(src[i]);
  }
}

// ---------------------------------------------------------------------------
// Generic batched WMMA GEMM.
//   C[m,n] (+)= sum_k A[m,k] * B(k,n)
//   BT=true : B stored as W[n,k] row-major (ldb strides n)  -> contiguous frag
//   BT=false: B stored as B[k,n] row-major (ldb strides k)  -> strided frag
//   OUT: 0 = fp32 store (+bias), 1 = fp32 accumulate, 2 = bf16 store (opt ReLU)
// One wave computes a 32(M) x 64(N) tile; block = 8 waves -> 256 M rows.
// REQUIRES: N % 64 == 0; K % 32 == 0; A/B rows padded so reads up to the
// next 32-row boundary are in-bounds.
// Batch z: off = (z/zDiv)*s?b + (z%zDiv)*s?h  (element offsets).
// ---------------------------------------------------------------------------
template <bool BT, int OUT, bool RELU>
__global__ __launch_bounds__(256) void gemm_wmma(
    const unsigned short* __restrict__ A, const unsigned short* __restrict__ Bm,
    float* __restrict__ Cf, unsigned short* __restrict__ Cb,
    const float* __restrict__ bias,
    int M, int N, int K, int lda, int ldb, int ldc, int mValid, int zDiv,
    long sAb, long sAh, long sBb, long sBh, long sCb, long sCh) {
  const int z = blockIdx.z;
  const long offA = (long)(z / zDiv) * sAb + (long)(z % zDiv) * sAh;
  const long offB = (long)(z / zDiv) * sBb + (long)(z % zDiv) * sBh;
  const long offC = (long)(z / zDiv) * sCb + (long)(z % zDiv) * sCh;

  const int wave = threadIdx.x >> 5;
  const int lane = threadIdx.x & 31;
  const int mbase = (blockIdx.y * 8 + wave) * 32;
  if (mbase >= M) return;                       // wave-uniform
  const int laneN = lane & 15;
  const bool hi = lane >= 16;
  const int n0base = blockIdx.x * 64;

  v8f acc[2][4];
  const v8f vz = {0.f, 0.f, 0.f, 0.f, 0.f, 0.f, 0.f, 0.f};
#pragma unroll
  for (int mi = 0; mi < 2; ++mi)
#pragma unroll
    for (int j = 0; j < 4; ++j) acc[mi][j] = vz;

  const int aoff = hi ? 16 : 0;
  const int boff = hi ? 16 : 0;
  const unsigned short* Arow0 = A + offA + (long)(mbase + laneN) * lda + aoff;
  const unsigned short* Arow1 = Arow0 + 16L * lda;
  const unsigned short* Bbase =
      BT ? (Bm + offB + (long)(n0base + laneN) * ldb + boff)
         : (Bm + offB + (long)boff * ldb + n0base + laneN);

#pragma unroll 2
  for (int k0 = 0; k0 < K; k0 += 32) {
    FragB16 fa0, fa1;
    fa0.q[0] = *(const uint4*)(Arow0 + k0);
    fa0.q[1] = *(const uint4*)(Arow0 + k0 + 8);
    fa1.q[0] = *(const uint4*)(Arow1 + k0);
    fa1.q[1] = *(const uint4*)(Arow1 + k0 + 8);
#pragma unroll
    for (int j = 0; j < 4; ++j) {
      FragB16 fb;
      if (BT) {
        const unsigned short* p = Bbase + (long)j * 16 * ldb + k0;
        fb.q[0] = *(const uint4*)(p);
        fb.q[1] = *(const uint4*)(p + 8);
      } else {
        const unsigned short* p = Bbase + (long)k0 * ldb + j * 16;
#pragma unroll
        for (int i = 0; i < 16; ++i) fb.s[i] = p[(long)i * ldb];
      }
      acc[0][j] = __builtin_amdgcn_wmma_f32_16x16x32_bf16(
          false, fa0.v, false, fb.v, (short)0, acc[0][j], false, false);
      acc[1][j] = __builtin_amdgcn_wmma_f32_16x16x32_bf16(
          false, fa1.v, false, fb.v, (short)0, acc[1][j], false, false);
    }
  }

#pragma unroll
  for (int mi = 0; mi < 2; ++mi) {
#pragma unroll
    for (int j = 0; j < 4; ++j) {
      const int n = n0base + j * 16 + laneN;
#pragma unroll
      for (int vr = 0; vr < 8; ++vr) {
        const int m = mbase + mi * 16 + vr + (hi ? 8 : 0);
        if (m >= mValid) continue;
        const long idx = offC + (long)m * ldc + n;
        float v = acc[mi][j][vr];
        if (OUT == 1) {
          Cf[idx] += v;
        } else if (OUT == 0) {
          if (bias) v += bias[n];
          Cf[idx] = v;
        } else {
          if (RELU) v = fmaxf(v, 0.f);
          Cb[idx] = f2bf(v);
        }
      }
    }
  }
}

// ---------------------------------------------------------------------------
// Assemble t[B, NP, TKN]: row0 = cls+pos, rows 1..196 = enc + b_enc + pos,
// pad rows zero. One block per (b, row), 256 threads x 3 channels.
// ---------------------------------------------------------------------------
__global__ void assemble_t(const float* __restrict__ enc,
                           const float* __restrict__ b_enc,
                           const float* __restrict__ cls,
                           const float* __restrict__ pos,
                           float* __restrict__ t) {
  const int row = blockIdx.x;               // b*NP + n
  const int b = row / NP, n = row % NP;
  float* out = t + (long)row * TKN;
#pragma unroll
  for (int e = 0; e < 3; ++e) {
    const int c = threadIdx.x + e * 256;
    float v;
    if (n == 0)      v = cls[c] + pos[c];
    else if (n < NB) v = enc[((long)b * N_ + (n - 1)) * TKN + c] + b_enc[c]
                         + pos[(long)n * TKN + c];
    else             v = 0.f;
    out[c] = v;
  }
}

// ---------------------------------------------------------------------------
// Energy LayerNorm: scalar gamma, per-channel bias; pad rows -> exact zeros.
// ---------------------------------------------------------------------------
__global__ void eln_kernel(const float* __restrict__ t,
                           unsigned short* __restrict__ g,
                           const float* __restrict__ gamma,
                           const float* __restrict__ bias) {
  __shared__ float red[256];
  const int row = blockIdx.x;
  const int n = row % NP;
  const int tid = threadIdx.x;
  unsigned short* out = g + (long)row * TKN;
  if (n >= NB) {                 // block-uniform
    out[tid] = 0; out[tid + 256] = 0; out[tid + 512] = 0;
    return;
  }
  const float* in = t + (long)row * TKN;
  float v0 = in[tid], v1 = in[tid + 256], v2 = in[tid + 512];
  red[tid] = v0 + v1 + v2; __syncthreads();
  for (int s = 128; s; s >>= 1) { if (tid < s) red[tid] += red[tid + s]; __syncthreads(); }
  const float mean = red[0] * (1.f / TKN); __syncthreads();
  const float x0 = v0 - mean, x1 = v1 - mean, x2 = v2 - mean;
  red[tid] = x0 * x0 + x1 * x1 + x2 * x2; __syncthreads();
  for (int s = 128; s; s >>= 1) { if (tid < s) red[tid] += red[tid + s]; __syncthreads(); }
  const float inv = rsqrtf(red[0] * (1.f / TKN) + EPS);
  const float gm = gamma[0];
  out[tid]       = f2bf(gm * x0 * inv + bias[tid]);
  out[tid + 256] = f2bf(gm * x1 * inv + bias[tid + 256]);
  out[tid + 512] = f2bf(gm * x2 * inv + bias[tid + 512]);
}

// Decoder LayerNorm: per-channel weight & bias (pad rows harmless, unused).
__global__ void ln_kernel(const float* __restrict__ t,
                          unsigned short* __restrict__ g,
                          const float* __restrict__ w,
                          const float* __restrict__ b) {
  __shared__ float red[256];
  const int row = blockIdx.x;
  const int tid = threadIdx.x;
  const float* in = t + (long)row * TKN;
  unsigned short* out = g + (long)row * TKN;
  float v0 = in[tid], v1 = in[tid + 256], v2 = in[tid + 512];
  red[tid] = v0 + v1 + v2; __syncthreads();
  for (int s = 128; s; s >>= 1) { if (tid < s) red[tid] += red[tid + s]; __syncthreads(); }
  const float mean = red[0] * (1.f / TKN); __syncthreads();
  const float x0 = v0 - mean, x1 = v1 - mean, x2 = v2 - mean;
  red[tid] = x0 * x0 + x1 * x1 + x2 * x2; __syncthreads();
  for (int s = 128; s; s >>= 1) { if (tid < s) red[tid] += red[tid + s]; __syncthreads(); }
  const float inv = rsqrtf(red[0] * (1.f / TKN) + EPS);
  out[tid]       = f2bf(x0 * inv * w[tid]       + b[tid]);
  out[tid + 256] = f2bf(x1 * inv * w[tid + 256] + b[tid + 256]);
  out[tid + 512] = f2bf(x2 * inv * w[tid + 512] + b[tid + 512]);
}

// ---------------------------------------------------------------------------
// Row softmax of beta*S over 197 valid keys -> P (bf16) and its transpose PT.
// S rows have leading dim SLD=256 (only first NB columns are meaningful).
// Pad rows/cols written as exact zeros. One block per (b,h,q) row; NP<=256.
// ---------------------------------------------------------------------------
__global__ void softmax_kernel(const float* __restrict__ S,
                               unsigned short* __restrict__ P,
                               unsigned short* __restrict__ PT) {
  __shared__ float red[256];
  const int idx = blockIdx.x;             // bh*NP + q
  const int q = idx % NP;
  const long bh = idx / NP;
  unsigned short* Prow = P + bh * (long)(NP * NP) + (long)q * NP;
  unsigned short* PTc  = PT + bh * (long)(NP * NP) + q;
  const int tid = threadIdx.x;
  if (q >= NB) {
    if (tid < NP) { Prow[tid] = 0; PTc[(long)tid * NP] = 0; }
    return;
  }
  const float* Srow = S + bh * (long)(NP * SLD) + (long)q * SLD;
  const float s = (tid < NB) ? Srow[tid] : -3.0e38f;
  red[tid] = s; __syncthreads();
  for (int k = 128; k; k >>= 1) { if (tid < k) red[tid] = fmaxf(red[tid], red[tid + k]); __syncthreads(); }
  const float mx = red[0]; __syncthreads();
  const float e = (tid < NB) ? __expf(BETA * (s - mx)) : 0.f;
  red[tid] = e; __syncthreads();
  for (int k = 128; k; k >>= 1) { if (tid < k) red[tid] += red[tid + k]; __syncthreads(); }
  const float inv = 1.f / red[0];
  if (tid < NP) {
    const unsigned short v = f2bf(e * inv);
    Prow[tid] = v;
    PTc[(long)tid * NP] = v;
  }
}

// ---------------------------------------------------------------------------
// Host launch
// ---------------------------------------------------------------------------
static inline dim3 gemm_grid(int M, int N, int Z) {
  return dim3((unsigned)((N + 63) / 64), (unsigned)((M + 255) / 256), (unsigned)Z);
}

extern "C" void kernel_launch(void* const* d_in, const int* in_sizes, int n_in,
                              void* d_out, int out_size, void* d_ws, size_t ws_size,
                              hipStream_t stream) {
  (void)in_sizes; (void)n_in; (void)out_size; (void)ws_size;
  const float* x     = (const float*)d_in[0];
  const float* w_enc = (const float*)d_in[1];
  const float* b_enc = (const float*)d_in[2];
  const float* cls   = (const float*)d_in[3];
  const float* pos   = (const float*)d_in[4];
  const float* eg    = (const float*)d_in[5];
  const float* eb    = (const float*)d_in[6];
  const float* wq    = (const float*)d_in[7];
  const float* wk    = (const float*)d_in[8];
  const float* whop  = (const float*)d_in[9];
  const float* dlnw  = (const float*)d_in[10];
  const float* dlnb  = (const float*)d_in[11];
  const float* w_dec = (const float*)d_in[12];
  const float* b_dec = (const float*)d_in[13];
  float* out = (float*)d_out;

  // ---- workspace carve-out (256B aligned) ----
  char* ws = (char*)d_ws;
  size_t off = 0;
  auto alloc = [&](size_t bytes) -> void* {
    void* p = ws + off;
    off = (off + bytes + 255) & ~(size_t)255;
    return p;
  };
  const long XE  = (long)B_ * N_ * TKN;       // encoder rows
  const long WE  = (long)TKN * TKN;
  const long WH  = (long)HID * TKN;
  const long MR  = (long)B_ * NP;             // padded token rows
  const long TE  = MR * TKN;
  const long SE  = (long)B_ * H * NP * SLD;   // fp32 scores, ld = 256
  const long PE  = (long)B_ * H * NP * NP;    // bf16 probabilities, ld = 224
  const long QKE = (MR + 32) * QKLD;          // [Q|K] + 32 rows slack
  const long ACE = MR * AC;                   // [U|V|r] combined activations

  unsigned short* xbf   = (unsigned short*)alloc(XE * 2);
  unsigned short* wencb = (unsigned short*)alloc(WE * 2);
  unsigned short* cqk   = (unsigned short*)alloc(2 * WE * 2);  // [wq ; wk]
  unsigned short* whb   = (unsigned short*)alloc(WH * 2);
  unsigned short* cw    = (unsigned short*)alloc((long)TKN * AC * 2); // [wq^T|wk^T|whop^T]
  unsigned short* wdecb = (unsigned short*)alloc(WE * 2);
  float*          enc   = (float*)alloc(XE * 4);
  float*          t     = (float*)alloc(TE * 4);
  unsigned short* g     = (unsigned short*)alloc((TE + 32L * TKN) * 2); // +slack (decode)
  unsigned short* QKb   = (unsigned short*)alloc(QKE * 2);
  float*          S     = (float*)alloc(SE * 4);
  unsigned short* P     = (unsigned short*)alloc(PE * 2);
  unsigned short* PT    = (unsigned short*)alloc(PE * 2);
  unsigned short* ACT   = (unsigned short*)alloc(ACE * 2);

  // ---- weight / input conversion to bf16 (each launch; deterministic) ----
  cvt_bf16<<<1024, 256, 0, stream>>>(x, xbf, XE);
  cvt_bf16<<<1024, 256, 0, stream>>>(w_enc, wencb, WE);
  cvt_bf16<<<1024, 256, 0, stream>>>(wq, cqk, WE);            // rows 0..767
  cvt_bf16<<<1024, 256, 0, stream>>>(wk, cqk + WE, WE);       // rows 768..1535
  cvt_bf16<<<1024, 256, 0, stream>>>(whop, whb, WH);
  cvt_bf16<<<1024, 256, 0, stream>>>(w_dec, wdecb, WE);
  cvt_bf16_t_ld<<<1024, 256, 0, stream>>>(wq, cw, TKN, TKN, AC, 0);
  cvt_bf16_t_ld<<<1024, 256, 0, stream>>>(wk, cw, TKN, TKN, AC, TKN);
  cvt_bf16_t_ld<<<1024, 256, 0, stream>>>(whop, cw, HID, TKN, AC, 2 * TKN);

  const long sQKb = (long)NP * QKLD;  // per-batch stride of [Q|K] buffer
  const long sACb = (long)NP * AC;    // per-batch stride of activation buffer
  const long sP   = (long)NP * NP;    // per-(b,h) stride of probability buffers
  const long sSc  = (long)NP * SLD;   // per-(b,h) stride of score buffer
  const unsigned short* Qv = QKb;           // Q view: cols 0..767
  const unsigned short* Kv = QKb + TKN;     // K view: cols 768..1535

  // ---- encode: enc = x @ w_enc^T ----
  gemm_wmma<true, 0, false><<<gemm_grid(B_ * N_, TKN, 1), 256, 0, stream>>>(
      xbf, wencb, enc, nullptr, nullptr,
      B_ * N_, TKN, D, D, D, TKN, B_ * N_, 1, 0, 0, 0, 0, 0, 0);
  assemble_t<<<B_ * NP, 256, 0, stream>>>(enc, b_enc, cls, pos, t);

  // ---- 12 energy-descent steps ----
  for (int step = 0; step < KST; ++step) {
    eln_kernel<<<B_ * NP, 256, 0, stream>>>(t, g, eg, eb);

    // [Q|K] = g @ [wq;wk]^T   (bf16 out, N = 1536)
    gemm_wmma<true, 2, false><<<gemm_grid((int)MR, QKLD, 1), 256, 0, stream>>>(
        g, cqk, nullptr, QKb, nullptr,
        (int)MR, QKLD, TKN, TKN, TKN, QKLD, (int)MR, 1, 0, 0, 0, 0, 0, 0);

    // S[b,h] = Q_bh @ K_bh^T  (fp32, N padded to 256, batched z = b*H + h)
    gemm_wmma<true, 0, false><<<gemm_grid(NP, SLD, B_ * H), 256, 0, stream>>>(
        Qv, Kv, S, nullptr, nullptr,
        NP, SLD, QK, QKLD, QKLD, SLD, NP,
        H, sQKb, 64, sQKb, 64, (long)H * sSc, sSc);

    softmax_kernel<<<B_ * H * NP, 256, 0, stream>>>(S, P, PT);

    // ACT[:, 0:768]    : U = P @ K     (bf16, N=64, B non-transposed)
    gemm_wmma<false, 2, false><<<gemm_grid(NP, QK, B_ * H), 256, 0, stream>>>(
        P, Kv, nullptr, ACT, nullptr,
        NP, QK, NP, NP, QKLD, AC, NP,
        H, (long)H * sP, sP, sQKb, 64, sACb, 64);
    // ACT[:, 768:1536] : V = P^T @ Q
    gemm_wmma<false, 2, false><<<gemm_grid(NP, QK, B_ * H), 256, 0, stream>>>(
        PT, Qv, nullptr, ACT + TKN, nullptr,
        NP, QK, NP, NP, QKLD, AC, NP,
        H, (long)H * sP, sP, sQKb, 64, sACb, 64);
    // ACT[:, 1536:4608]: r = relu(g @ w_hop^T)   (bf16, N=3072)
    gemm_wmma<true, 2, true><<<gemm_grid((int)MR, HID, 1), 256, 0, stream>>>(
        g, whb, nullptr, ACT + 2 * TKN, nullptr,
        (int)MR, HID, TKN, TKN, TKN, AC, (int)MR, 1, 0, 0, 0, 0, 0, 0);

    // t += ACT @ CW^T   (single fused fp32 accumulate, K = 4608)
    gemm_wmma<true, 1, false><<<gemm_grid((int)MR, TKN, 1), 256, 0, stream>>>(
        ACT, cw, t, nullptr, nullptr,
        (int)MR, TKN, AC, AC, AC, TKN, (int)MR, 1, 0, 0, 0, 0, 0, 0);
  }

  // ---- decode: LN then y = LN(t)[:,1:] @ w_dec^T + b_dec ----
  ln_kernel<<<B_ * NP, 256, 0, stream>>>(t, g, dlnw, dlnb);
  gemm_wmma<true, 0, false><<<gemm_grid(N_, D, B_), 256, 0, stream>>>(
      g + TKN /* skip cls row */, wdecb, out, nullptr, b_dec,
      N_, D, TKN, TKN, TKN, D, N_,
      1, (long)NP * TKN, 0, 0, 0, (long)N_ * D, 0);
}
